// MeshNet_45878840656272
// MI455X (gfx1250) — matrix operand order repaired and checked by reading, not compiled, not run on hardware
//
#include <hip/hip_runtime.h>
#include <hip/hip_bf16.h>

// ---------------------------------------------------------------------------
// MeshNet forward for MI455X (gfx1250).
// Memory-bound (~0.5 TFLOP GEMM vs ~1 GB f16 activation traffic @ 23.3 TB/s)
// => f16 activations, v_wmma_f32_16x16x32_f16 GEMMs with fused BN/ReLU,
//    concat-free multi-source K-tiles, gather-fused mesh-conv with
//    max-over-3-neighbors in registers.
// Round 6: fix host-pass narrowing cast in the async LDS-offset computation
// (guard asm with __HIP_DEVICE_COMPILE__, cast via u64). A-tile staging via
// GLOBAL_LOAD_ASYNC_TO_LDS_B128 (ASYNCcnt), s_wait_asynccnt 0 before barrier.
// ---------------------------------------------------------------------------

typedef __attribute__((ext_vector_type(16))) _Float16 v16h;
typedef __attribute__((ext_vector_type(8)))  _Float16 v8h;
typedef __attribute__((ext_vector_type(4)))  _Float16 v4h;
typedef __attribute__((ext_vector_type(8)))  float    v8f;
typedef __attribute__((ext_vector_type(4)))  float    v4f;

typedef __attribute__((address_space(3))) const void* lds_cvptr;

constexpr int BATCH = 8;
constexpr int NPTS  = 8192;
constexpr int NKK   = 64;
constexpr int MAXK  = 1792;   // largest padded K (concat head)

#define USE_ASYNC_A 1   // stage A tile with global_load_async_to_lds_b128

#if defined(__has_builtin)
#  if __has_builtin(__builtin_amdgcn_s_wait_tensorcnt)
#    define TENSOR_WAIT0() __builtin_amdgcn_s_wait_tensorcnt(0)
#  endif
#endif
#ifndef TENSOR_WAIT0
#  define TENSOR_WAIT0() ((void)0)
#endif

// Up to 3 concatenated channel-range sources, as u32 byte-offsets from the
// workspace base (all activations live in d_ws; total < 4 GB).
struct Cat3 {
  unsigned o0, o1, o2;   // source base byte-offsets in ws
  unsigned s0, s1, s2;   // per-batch byte strides
  int      c0, c1, c2;   // channel counts
};

__device__ __forceinline__ unsigned
cat_off(const Cat3& X, int b, int cin) {
  if (cin < X.c0) return X.o0 + (unsigned)b * X.s0 + (unsigned)cin * (NPTS * 2);
  cin -= X.c0;
  if (cin < X.c1) return X.o1 + (unsigned)b * X.s1 + (unsigned)cin * (NPTS * 2);
  cin -= X.c1;
  return X.o2 + (unsigned)b * X.s2 + (unsigned)cin * (NPTS * 2);
}

__device__ __forceinline__ v16h cat16(v8h lo, v8h hi) {
  return __builtin_shufflevector(lo, hi, 0, 1, 2, 3, 4, 5, 6, 7,
                                 8, 9, 10, 11, 12, 13, 14, 15);
}

// Async 16B global->LDS copy (GLOBAL_LOAD_ASYNC_TO_LDS_B128, tracked by
// ASYNCcnt). dst must be a generic pointer to __shared__ memory.
__device__ __forceinline__ void async_g2l_b128(void* lds_dst, const void* gsrc) {
#if USE_ASYNC_A && defined(__HIP_DEVICE_COMPILE__)
  // addrspacecast -> 32-bit LDS offset (AS(3) pointers are 32-bit on device)
  unsigned loff = (unsigned)(unsigned long long)(lds_cvptr)lds_dst;
  asm volatile("global_load_async_to_lds_b128 %0, %1, off"
               :: "v"(loff), "v"(gsrc) : "memory");
#else
  *(v8h*)lds_dst = *(const v8h*)gsrc;
#endif
}

__device__ __forceinline__ void async_wait0() {
#if USE_ASYNC_A && defined(__HIP_DEVICE_COMPILE__)
  asm volatile("s_wait_asynccnt 0" ::: "memory");
#endif
}

// ---------------------------------------------------------------------------
// Conv1x1 + BN + ReLU via WMMA.
// Block: 256 threads = 8 waves; block tile 32 cout x 128 n; K step 64.
// Wave w: cout sub-tile (w>>2), n sub-tiles (w&3), (w&3)+4 (shared A frag).
// LDS: As[cout][k] / BsT[n][k], stride 72 halves => every fragment is a
// contiguous 16B-aligned ds_load_b128 (k = 16*(lane/16) + s in both A and B, wave32).
// ptab[k]: u32 ws-offset of the concat-source row (built once per block).
// Weights pre-padded: Cout->%32, Cin->%64, zero-filled => no staging guards.
// ---------------------------------------------------------------------------
__global__ __launch_bounds__(256)
void gemm_cbr_kernel(const char* __restrict__ WSB, Cat3 X,
                     const _Float16* __restrict__ W,
                     const float* __restrict__ bias, const float* __restrict__ sc,
                     const float* __restrict__ sh, _Float16* __restrict__ Y,
                     int Cout, int Cin, int Cinp) {
  __shared__ _Float16 As[32][72];
  __shared__ _Float16 BsT[128][72];
  __shared__ unsigned ptab[MAXK];
  const int tid = threadIdx.x;
  const int b  = blockIdx.z;
  const int m0 = blockIdx.y * 32;
  const int n0 = blockIdx.x * 128;
  const int lane = tid & 31, wv = tid >> 5;
  const int ct = wv >> 2, nt = wv & 3;
  const int hi = lane >> 4, lo = lane & 15;
  const int ar = tid >> 3, ac = (tid & 7) * 8;        // A: row, 8 halves
  const int br = (tid >> 5) * 8, bc = (tid & 31) * 4; // B: 8 rows x 4 cols

  TENSOR_WAIT0();
  // one-time divergent source select + clamp, amortized over all K-steps
  for (int i = tid; i < Cinp; i += 256)
    ptab[i] = cat_off(X, b, (i < Cin) ? i : (Cin - 1)) + (unsigned)(n0 * 2);
  __syncthreads();

  v8f acc0 = {}, acc1 = {};
  const int arow = ct * 16 + lo;
  const int col0 = nt * 16 + lo, col1 = (nt + 4) * 16 + lo;
  for (int k0 = 0; k0 < Cinp; k0 += 64) {
    // A: async DMA of the 4KB weight tile straight into LDS (ASYNCcnt)
    async_g2l_b128(&As[ar][ac], &W[(long)(m0 + ar) * Cinp + k0 + ac]);
    // B: u32 offset from LDS table -> global b64 loads; 8x4 micro-transpose
    {
      v4h rv[8];
#pragma unroll
      for (int rr = 0; rr < 8; ++rr)
        rv[rr] = *(const v4h*)(WSB + (ptab[k0 + br + rr] + (unsigned)(bc * 2)));
#pragma unroll
      for (int i = 0; i < 4; ++i) {
        v8h cv = {rv[0][i], rv[1][i], rv[2][i], rv[3][i],
                  rv[4][i], rv[5][i], rv[6][i], rv[7][i]};
        *(v8h*)&BsT[bc + i][br] = cv;
      }
    }
    async_wait0();          // ASYNCcnt not covered by the barrier
    __syncthreads();
    if (k0 + 64 < Cinp)
      __builtin_prefetch(&W[(long)(m0 + ar) * Cinp + k0 + 64], 0, 0);
#pragma unroll
    for (int kh = 0; kh < 2; ++kh) {
      const int kb = kh * 32;
      v16h a = cat16(*(const v8h*)&As[arow][kb + hi * 16],
                     *(const v8h*)&As[arow][kb + hi * 16 + 8]);
      v16h bt0 = cat16(*(const v8h*)&BsT[col0][kb + hi * 16],
                       *(const v8h*)&BsT[col0][kb + hi * 16 + 8]);
      acc0 = __builtin_amdgcn_wmma_f32_16x16x32_f16(false, a, false, bt0,
                                                    (short)0, acc0, false, false);
      v16h bt1 = cat16(*(const v8h*)&BsT[col1][kb + hi * 16],
                       *(const v8h*)&BsT[col1][kb + hi * 16 + 8]);
      acc1 = __builtin_amdgcn_wmma_f32_16x16x32_f16(false, a, false, bt1,
                                                    (short)0, acc1, false, false);
    }
    __syncthreads();
  }
  // fused bias + folded-BN + ReLU epilogue; f16 store
#pragma unroll
  for (int r = 0; r < 8; ++r) {
    int m = m0 + ct * 16 + hi * 8 + r;
    if (m < Cout) {
      float bm = bias[m], sm = sc[m], hm = sh[m];
      long base = ((long)b * Cout + m) * NPTS + n0 + lo;
      Y[base + nt * 16]       = (_Float16)fmaxf((acc0[r] + bm) * sm + hm, 0.f);
      Y[base + (nt + 4) * 16] = (_Float16)fmaxf((acc1[r] + bm) * sm + hm, 0.f);
    }
  }
}

// ---------------------------------------------------------------------------
// Mesh-conv "Concat" aggregation, fully fused:
//   Y[cout,n] = max_{j=0..2} relu(BN(W * [st[:,n]; st[:,idx[n,j]]] + b))
// Center rows (k < C) stream vectorized; gathered rows fetch per-column via
// u32 ws-offsets (global path, base+voffset addressing).
// ---------------------------------------------------------------------------
__global__ __launch_bounds__(256)
void gemm_cbr_nb_kernel(const char* __restrict__ WSB, unsigned stOff, int C,
                        const int* __restrict__ nbidx,
                        const _Float16* __restrict__ W,
                        const float* __restrict__ bias, const float* __restrict__ sc,
                        const float* __restrict__ sh,
                        _Float16* __restrict__ Y, int Cout, int Cinp) {
  __shared__ _Float16 As[32][72];
  __shared__ _Float16 BsT[128][72];
  __shared__ unsigned rtab[512];
  __shared__ int idxs[128 * 3];
  const int tid = threadIdx.x;
  const int b = blockIdx.z, m0 = blockIdx.y * 32, n0 = blockIdx.x * 128;
  const int lane = tid & 31, wv = tid >> 5;
  const int ct = wv >> 2, nt = wv & 3, hi = lane >> 4, lo = lane & 15;
  const int ar = tid >> 3, ac = (tid & 7) * 8;
  const int br = (tid >> 5) * 8, bc = (tid & 31) * 4;
  const int Cin = 2 * C;

  const unsigned stB = stOff + (unsigned)b * (unsigned)(C * NPTS * 2);
  for (int i = tid; i < 128 * 3; i += 256)
    idxs[i] = nbidx[((long)b * NPTS + n0 + i / 3) * 3 + (i % 3)];
  for (int i = tid; i < Cinp; i += 256) {
    int kc = (i < Cin) ? i : (Cin - 1);
    rtab[i] = stB + (unsigned)((kc < C) ? kc : (kc - C)) * (NPTS * 2);
  }
  __syncthreads();

  float mx0[8] = {0.f, 0.f, 0.f, 0.f, 0.f, 0.f, 0.f, 0.f};  // relu >= 0
  float mx1[8] = {0.f, 0.f, 0.f, 0.f, 0.f, 0.f, 0.f, 0.f};
  const int arow = ct * 16 + lo;
  const int col0 = nt * 16 + lo, col1 = (nt + 4) * 16 + lo;

  for (int j = 0; j < 3; ++j) {
    v8f acc0 = {}, acc1 = {};
    for (int k0 = 0; k0 < Cinp; k0 += 64) {
      async_g2l_b128(&As[ar][ac], &W[(long)(m0 + ar) * Cinp + k0 + ac]);
      {
        v4h rv[8];
#pragma unroll
        for (int rr = 0; rr < 8; ++rr) {
          int kk = k0 + br + rr;
          int kc = (kk < Cin) ? kk : (Cin - 1);
          unsigned ro = rtab[kk];
          if (kc < C) {
            rv[rr] = *(const v4h*)(WSB + (ro + (unsigned)((n0 + bc) * 2)));
          } else {
            v4h g;
            g[0] = *(const _Float16*)(WSB + (ro + ((unsigned)idxs[(bc + 0) * 3 + j] << 1)));
            g[1] = *(const _Float16*)(WSB + (ro + ((unsigned)idxs[(bc + 1) * 3 + j] << 1)));
            g[2] = *(const _Float16*)(WSB + (ro + ((unsigned)idxs[(bc + 2) * 3 + j] << 1)));
            g[3] = *(const _Float16*)(WSB + (ro + ((unsigned)idxs[(bc + 3) * 3 + j] << 1)));
            rv[rr] = g;
          }
        }
#pragma unroll
        for (int i = 0; i < 4; ++i) {
          v8h cv = {rv[0][i], rv[1][i], rv[2][i], rv[3][i],
                    rv[4][i], rv[5][i], rv[6][i], rv[7][i]};
          *(v8h*)&BsT[bc + i][br] = cv;
        }
      }
      async_wait0();
      __syncthreads();
#pragma unroll
      for (int kh = 0; kh < 2; ++kh) {
        const int kb = kh * 32;
        v16h a = cat16(*(const v8h*)&As[arow][kb + hi * 16],
                       *(const v8h*)&As[arow][kb + hi * 16 + 8]);
        v16h bt0 = cat16(*(const v8h*)&BsT[col0][kb + hi * 16],
                         *(const v8h*)&BsT[col0][kb + hi * 16 + 8]);
        acc0 = __builtin_amdgcn_wmma_f32_16x16x32_f16(false, a, false, bt0,
                                                      (short)0, acc0, false, false);
        v16h bt1 = cat16(*(const v8h*)&BsT[col1][kb + hi * 16],
                         *(const v8h*)&BsT[col1][kb + hi * 16 + 8]);
        acc1 = __builtin_amdgcn_wmma_f32_16x16x32_f16(false, a, false, bt1,
                                                      (short)0, acc1, false, false);
      }
      __syncthreads();
    }
#pragma unroll
    for (int r = 0; r < 8; ++r) {
      int m = m0 + ct * 16 + hi * 8 + r;
      if (m < Cout) {
        float bm = bias[m], sm = sc[m], hm = sh[m];
        mx0[r] = fmaxf(mx0[r], fmaxf((acc0[r] + bm) * sm + hm, 0.f));
        mx1[r] = fmaxf(mx1[r], fmaxf((acc1[r] + bm) * sm + hm, 0.f));
      }
    }
  }
#pragma unroll
  for (int r = 0; r < 8; ++r) {
    int m = m0 + ct * 16 + hi * 8 + r;
    if (m < Cout) {
      long base = ((long)b * Cout + m) * NPTS + n0 + lo;
      Y[base + nt * 16]       = (_Float16)mx0[r];
      Y[base + (nt + 4) * 16] = (_Float16)mx1[r];
    }
  }
}

// ---------------------------------------------------------------------------
// Face Kernel Correlation: 64 kernels x 4 dirs cached in LDS, 1024 exp/point.
// ---------------------------------------------------------------------------
__global__ __launch_bounds__(256)
void fkc_kernel(const float* __restrict__ normals, const int* __restrict__ nbidx,
                const float* __restrict__ alpha, const float* __restrict__ beta,
                const float* __restrict__ s, const float* __restrict__ sb,
                _Float16* __restrict__ out) {
  __shared__ float wx[256], wy[256], wz[256], w2[256];
  const int tid = threadIdx.x, b = blockIdx.y;
  {
    float a = alpha[tid], be = beta[tid];
    float sa = __sinf(a);
    float x = sa * __cosf(be), y = sa * __sinf(be), z = __cosf(a);
    wx[tid] = x; wy[tid] = y; wz[tid] = z; w2[tid] = x * x + y * y + z * z;
  }
  __syncthreads();
  const int n = blockIdx.x * 256 + tid;
  const float* nb = normals + (long)b * 3 * NPTS;
  float fx[4], fy[4], fz[4], f2[4];
  fx[0] = nb[n]; fy[0] = nb[NPTS + n]; fz[0] = nb[2 * NPTS + n];
  for (int t = 0; t < 3; ++t) {
    int nn = nbidx[((long)b * NPTS + n) * 3 + t];
    fx[t + 1] = nb[nn]; fy[t + 1] = nb[NPTS + nn]; fz[t + 1] = nb[2 * NPTS + nn];
  }
#pragma unroll
  for (int i = 0; i < 4; ++i) f2[i] = fx[i] * fx[i] + fy[i] * fy[i] + fz[i] * fz[i];
  for (int k = 0; k < NKK; ++k) {
    float acc = 0.f;
#pragma unroll
    for (int j = 0; j < 4; ++j) {
      int kj = k * 4 + j;
      float xx = wx[kj], yy = wy[kj], zz = wz[kj], ww = w2[kj];
#pragma unroll
      for (int i = 0; i < 4; ++i) {
        float cr = fx[i] * xx + fy[i] * yy + fz[i] * zz;
        float d = f2[i] + ww - 2.f * cr;
        acc += __expf(d * -12.5f);  // 1/(-2*sigma^2), sigma=0.2
      }
    }
    float yv = fmaxf((acc * (1.f / 16.f)) * s[k] + sb[k], 0.f);
    out[((long)b * NKK + k) * NPTS + n] = (_Float16)yv;
  }
}

// ------------------------- small utility kernels ---------------------------
__global__ void cvt4_f32_f16_kernel(const float* __restrict__ in,
                                    _Float16* __restrict__ out, long n4) {
  long i = (long)blockIdx.x * blockDim.x + threadIdx.x;
  if (i < n4) {
    v4f v = *(const v4f*)&in[i * 4];
    v4h h = {(_Float16)v[0], (_Float16)v[1], (_Float16)v[2], (_Float16)v[3]};
    *(v4h*)&out[i * 4] = h;
  }
}

// f32 [Cout][Cin] -> f16 [Coutp][Cinp], zero-padded both dims (guard-free
// WMMA A staging).
__global__ void cvt_pad_kernel(const float* __restrict__ in, _Float16* __restrict__ out,
                               int Cout, int Cin, int Coutp, int Cinp) {
  long i = (long)blockIdx.x * blockDim.x + threadIdx.x;
  if (i >= (long)Coutp * Cinp) return;
  int m = (int)(i / Cinp), k = (int)(i - (long)m * Cinp);
  out[i] = (m < Cout && k < Cin) ? (_Float16)in[(long)m * Cin + k] : (_Float16)0.f;
}

__global__ void avg3_kernel(const _Float16* __restrict__ a, const _Float16* __restrict__ b,
                            const _Float16* __restrict__ c, _Float16* __restrict__ o, long n) {
  long i = (long)blockIdx.x * blockDim.x + threadIdx.x;
  if (i < n)
    o[i] = (_Float16)(((float)a[i] + (float)b[i] + (float)c[i]) * (1.f / 3.f));
}

__global__ __launch_bounds__(256)
void maxn_kernel(const _Float16* __restrict__ in, float* __restrict__ out, int C) {
  __shared__ float red[256];
  const int c = blockIdx.x, b = blockIdx.y, tid = threadIdx.x;
  const _Float16* p = in + ((long)b * C + c) * NPTS;
  float m = -3.0e38f;
  for (int i = tid; i < NPTS; i += 256) m = fmaxf(m, (float)p[i]);
  red[tid] = m; __syncthreads();
  for (int sft = 128; sft > 0; sft >>= 1) {
    if (tid < sft) red[tid] = fmaxf(red[tid], red[tid + sft]);
    __syncthreads();
  }
  if (tid == 0) out[(long)b * C + c] = red[0];
}

__global__ void linear_kernel(const float* __restrict__ x, const float* __restrict__ w,
                              const float* __restrict__ bias, float* __restrict__ y,
                              int I, int O, int relu) {
  int t = blockIdx.x * blockDim.x + threadIdx.x;
  if (t >= BATCH * O) return;
  int b = t / O, o = t % O;
  const float* xr = x + (long)b * I;
  const float* wr = w + (long)o * I;
  float acc = bias[o];
  for (int i = 0; i < I; ++i) acc += xr[i] * wr[i];
  if (relu) acc = fmaxf(acc, 0.f);
  y[t] = acc;
}

// ---------------------------------------------------------------------------
// d_in layout: setup_inputs() order; params pytree flattened alphabetically
// (JAX dict ordering), leaves of each _cbr in order b, s, sb, w.
// ---------------------------------------------------------------------------
enum {
  IN_CENTERS = 0, IN_CORNERS, IN_NORMALS, IN_NBIDX,
  P_CONCAT_B, P_CONCAT_S, P_CONCAT_SB, P_CONCAT_W,
  P_FKC_ALPHA, P_FKC_BETA, P_FKC_S, P_FKC_SB,
  P_FUS1_B, P_FUS1_S, P_FUS1_SB, P_FUS1_W,
  P_FUS2_B, P_FUS2_S, P_FUS2_SB, P_FUS2_W,
  P_ROT1_B, P_ROT1_S, P_ROT1_SB, P_ROT1_W,
  P_ROT2_B, P_ROT2_S, P_ROT2_SB, P_ROT2_W,
  P_FUSION_B, P_FUSION_S, P_FUSION_SB, P_FUSION_W,
  P_LIN1_B, P_LIN1_W, P_LIN2_B, P_LIN2_W, P_LIN3_B, P_LIN3_W,
  P_MC1AGG_B, P_MC1AGG_S, P_MC1AGG_SB, P_MC1AGG_W,
  P_MC1CAT_B, P_MC1CAT_S, P_MC1CAT_SB, P_MC1CAT_W,
  P_MC1COMB_B, P_MC1COMB_S, P_MC1COMB_SB, P_MC1COMB_W,
  P_MC2AGG_B, P_MC2AGG_S, P_MC2AGG_SB, P_MC2AGG_W,
  P_MC2CAT_B, P_MC2CAT_S, P_MC2CAT_SB, P_MC2CAT_W,
  P_MC2COMB_B, P_MC2COMB_S, P_MC2COMB_SB, P_MC2COMB_W,
  P_SPAT1_B, P_SPAT1_S, P_SPAT1_SB, P_SPAT1_W,
  P_SPAT2_B, P_SPAT2_S, P_SPAT2_SB, P_SPAT2_W,
  P_STR1_B, P_STR1_S, P_STR1_SB, P_STR1_W,
  P_STR2_B, P_STR2_S, P_STR2_SB, P_STR2_W
};

struct SrcR { unsigned o; unsigned bs; int c; bool valid; };
static inline int padkk(int c) { return (c + 63) & ~63; }   // K pad (step 64)
static inline int padmm(int c) { return (c + 31) & ~31; }   // M pad

extern "C" void kernel_launch(void* const* d_in, const int* in_sizes, int n_in,
                              void* d_out, int out_size, void* d_ws, size_t ws_size,
                              hipStream_t stream) {
  (void)in_sizes; (void)n_in; (void)out_size; (void)ws_size;
  const float* centers = (const float*)d_in[IN_CENTERS];
  const float* corners = (const float*)d_in[IN_CORNERS];
  const float* normals = (const float*)d_in[IN_NORMALS];
  const int*   nbidx   = (const int*)d_in[IN_NBIDX];
  float* out = (float*)d_out;
  const char* WSB = (const char*)d_ws;

  size_t off = 0;
  auto alloc = [&](size_t bytes) -> void* {
    void* p = (char*)d_ws + off;
    off = (off + bytes + 255) & ~(size_t)255;
    return p;
  };
  auto wsoff = [&](const void* p) -> unsigned {
    return (unsigned)((const char*)p - WSB);
  };
  auto actbuf = [&](int C) -> _Float16* {
    return (_Float16*)alloc((size_t)BATCH * C * NPTS * sizeof(_Float16));
  };
  auto cvt = [&](const float* src, long n) -> const _Float16* {
    _Float16* dst = (_Float16*)alloc((size_t)n * sizeof(_Float16));
    long n4 = n / 4;
    cvt4_f32_f16_kernel<<<dim3((unsigned)((n4 + 255) / 256)), 256, 0, stream>>>(src, dst, n4);
    return dst;
  };
  auto cvtw = [&](int widx, int Cout, int Cin) -> const _Float16* {
    int Coutp = padmm(Cout), Cinp = padkk(Cin);
    long n = (long)Coutp * Cinp;
    _Float16* dst = (_Float16*)alloc((size_t)n * sizeof(_Float16));
    cvt_pad_kernel<<<dim3((unsigned)((n + 255) / 256)), 256, 0, stream>>>(
        (const float*)d_in[widx], dst, Cout, Cin, Coutp, Cinp);
    return dst;
  };
  auto cat = [](SrcR a, SrcR b2, SrcR c2) -> Cat3 {
    Cat3 x;
    x.o0 = a.o; x.s0 = a.bs; x.c0 = a.c;
    x.o1 = b2.valid ? b2.o : a.o; x.s1 = b2.bs; x.c1 = b2.c;
    x.o2 = c2.valid ? c2.o : a.o; x.s2 = c2.bs; x.c2 = c2.c;
    return x;
  };
  const SrcR none{0, 0, 0, false};
  auto full = [&](const _Float16* p, int C) -> SrcR {
    return SrcR{wsoff(p), (unsigned)(C * NPTS * 2), C, true};
  };
  auto sub = [&](const _Float16* base, int chOff, int bufC, int C) -> SrcR {
    return SrcR{wsoff(base) + (unsigned)(chOff * NPTS * 2),
                (unsigned)(bufC * NPTS * 2), C, true};
  };
  auto gemm = [&](Cat3 X, const _Float16* W, int ib, int is, int isb,
                  _Float16* Y, int Cout, int Cin) {
    dim3 g(NPTS / 128, (Cout + 31) / 32, BATCH);
    gemm_cbr_kernel<<<g, 256, 0, stream>>>(
        WSB, X, W, (const float*)d_in[ib], (const float*)d_in[is],
        (const float*)d_in[isb], Y, Cout, Cin, padkk(Cin));
  };

  // f16 copies of raw geometry inputs
  const _Float16* h_cent = cvt(centers, (long)BATCH * 3 * NPTS);
  const _Float16* h_corn = cvt(corners, (long)BATCH * 9 * NPTS);
  const _Float16* h_norm = cvt(normals, (long)BATCH * 3 * NPTS);

  // f16 weights, padded Cout %32 / Cin %64 (converted once, ~8 MB total)
  const _Float16* w_spat1 = cvtw(P_SPAT1_W, 64, 3);
  const _Float16* w_spat2 = cvtw(P_SPAT2_W, 64, 64);
  const _Float16* w_rot1  = cvtw(P_ROT1_W, 32, 6);
  const _Float16* w_rot2  = cvtw(P_ROT2_W, 32, 32);
  const _Float16* w_fus1  = cvtw(P_FUS1_W, 64, 32);
  const _Float16* w_fus2  = cvtw(P_FUS2_W, 64, 64);
  const _Float16* w_str1  = cvtw(P_STR1_W, 131, 131);
  const _Float16* w_str2  = cvtw(P_STR2_W, 131, 131);
  const _Float16* w_m1co  = cvtw(P_MC1COMB_W, 256, 195);
  const _Float16* w_m1ca  = cvtw(P_MC1CAT_W, 131, 262);
  const _Float16* w_m1ag  = cvtw(P_MC1AGG_W, 256, 131);
  const _Float16* w_m2co  = cvtw(P_MC2COMB_W, 512, 512);
  const _Float16* w_m2ca  = cvtw(P_MC2CAT_W, 256, 512);
  const _Float16* w_m2ag  = cvtw(P_MC2AGG_W, 512, 256);
  const _Float16* w_fusn  = cvtw(P_FUSION_W, 1024, 1024);
  const _Float16* w_conc  = cvtw(P_CONCAT_W, 1024, 1792);

  // ---- SpatialDescriptor ----
  _Float16* sp_t = actbuf(64);
  gemm(cat(full(h_cent, 3), none, none), w_spat1, P_SPAT1_B, P_SPAT1_S, P_SPAT1_SB, sp_t, 64, 3);
  _Float16* sp0 = actbuf(64);
  gemm(cat(full(sp_t, 64), none, none), w_spat2, P_SPAT2_B, P_SPAT2_S, P_SPAT2_SB, sp0, 64, 64);

  // ---- FRC (rotation-invariant corner branch) ----
  _Float16* rt = actbuf(32);
  _Float16* x1 = actbuf(32); _Float16* x2 = actbuf(32); _Float16* x3 = actbuf(32);
  gemm(cat(sub(h_corn, 0, 9, 6), none, none), w_rot1, P_ROT1_B, P_ROT1_S, P_ROT1_SB, rt, 32, 6);
  gemm(cat(full(rt, 32), none, none), w_rot2, P_ROT2_B, P_ROT2_S, P_ROT2_SB, x1, 32, 32);
  gemm(cat(sub(h_corn, 3, 9, 6), none, none), w_rot1, P_ROT1_B, P_ROT1_S, P_ROT1_SB, rt, 32, 6);
  gemm(cat(full(rt, 32), none, none), w_rot2, P_ROT2_B, P_ROT2_S, P_ROT2_SB, x2, 32, 32);
  gemm(cat(sub(h_corn, 6, 9, 3), sub(h_corn, 0, 9, 3), none),
       w_rot1, P_ROT1_B, P_ROT1_S, P_ROT1_SB, rt, 32, 6);
  gemm(cat(full(rt, 32), none, none), w_rot2, P_ROT2_B, P_ROT2_S, P_ROT2_SB, x3, 32, 32);
  _Float16* fea32 = actbuf(32);
  {
    long n = (long)BATCH * 32 * NPTS;
    avg3_kernel<<<dim3((unsigned)((n + 255) / 256)), 256, 0, stream>>>(x1, x2, x3, fea32, n);
  }
  _Float16* f64 = actbuf(64);
  gemm(cat(full(fea32, 32), none, none), w_fus1, P_FUS1_B, P_FUS1_S, P_FUS1_SB, f64, 64, 32);
  _Float16* s1 = actbuf(64);
  gemm(cat(full(f64, 64), none, none), w_fus2, P_FUS2_B, P_FUS2_S, P_FUS2_SB, s1, 64, 64);

  // ---- FKC ----
  _Float16* s2 = actbuf(NKK);
  fkc_kernel<<<dim3(NPTS / 256, BATCH), 256, 0, stream>>>(
      normals, nbidx, (const float*)d_in[P_FKC_ALPHA], (const float*)d_in[P_FKC_BETA],
      (const float*)d_in[P_FKC_S], (const float*)d_in[P_FKC_SB], s2);

  // ---- StructuralDescriptor ----
  _Float16* st_t = actbuf(131);
  gemm(cat(full(s1, 64), full(s2, 64), full(h_norm, 3)), w_str1,
       P_STR1_B, P_STR1_S, P_STR1_SB, st_t, 131, 131);
  _Float16* st0 = actbuf(131);
  gemm(cat(full(st_t, 131), none, none), w_str2, P_STR2_B, P_STR2_S, P_STR2_SB, st0, 131, 131);

  // ---- MeshConv 1 ----
  _Float16* sp1 = actbuf(256);
  gemm(cat(full(sp0, 64), full(st0, 131), none), w_m1co,
       P_MC1COMB_B, P_MC1COMB_S, P_MC1COMB_SB, sp1, 256, 195);
  _Float16* st1p = actbuf(131);
  gemm_cbr_nb_kernel<<<dim3(NPTS / 128, (131 + 31) / 32, BATCH), 256, 0, stream>>>(
      WSB, wsoff(st0), 131, nbidx, w_m1ca, (const float*)d_in[P_MC1CAT_B],
      (const float*)d_in[P_MC1CAT_S], (const float*)d_in[P_MC1CAT_SB], st1p, 131, padkk(262));
  _Float16* st1 = actbuf(256);
  gemm(cat(full(st1p, 131), none, none), w_m1ag, P_MC1AGG_B, P_MC1AGG_S, P_MC1AGG_SB, st1, 256, 131);

  // ---- MeshConv 2 ----
  _Float16* sp2 = actbuf(512);
  gemm(cat(full(sp1, 256), full(st1, 256), none), w_m2co,
       P_MC2COMB_B, P_MC2COMB_S, P_MC2COMB_SB, sp2, 512, 512);
  _Float16* st2p = actbuf(256);
  gemm_cbr_nb_kernel<<<dim3(NPTS / 128, (256 + 31) / 32, BATCH), 256, 0, stream>>>(
      WSB, wsoff(st1), 256, nbidx, w_m2ca, (const float*)d_in[P_MC2CAT_B],
      (const float*)d_in[P_MC2CAT_S], (const float*)d_in[P_MC2CAT_SB], st2p, 256, padkk(512));
  _Float16* st2 = actbuf(512);
  gemm(cat(full(st2p, 256), none, none), w_m2ag, P_MC2AGG_B, P_MC2AGG_S, P_MC2AGG_SB, st2, 512, 256);

  // ---- fusion + concat head ----
  _Float16* sp3 = actbuf(1024);
  gemm(cat(full(sp2, 512), full(st2, 512), none), w_fusn,
       P_FUSION_B, P_FUSION_S, P_FUSION_SB, sp3, 1024, 1024);
  _Float16* feaC = actbuf(1024);
  gemm(cat(full(sp1, 256), full(sp2, 512), full(sp3, 1024)), w_conc,
       P_CONCAT_B, P_CONCAT_S, P_CONCAT_SB, feaC, 1024, 1792);

  // ---- global max pool + classifier ----
  float* maxfea = (float*)alloc((size_t)BATCH * 1024 * sizeof(float));
  maxn_kernel<<<dim3(1024, BATCH), 256, 0, stream>>>(feaC, maxfea, 1024);
  float* t512 = (float*)alloc((size_t)BATCH * 512 * sizeof(float));
  float* t256 = (float*)alloc((size_t)BATCH * 256 * sizeof(float));
  linear_kernel<<<dim3((BATCH * 512 + 255) / 256), 256, 0, stream>>>(
      maxfea, (const float*)d_in[P_LIN1_W], (const float*)d_in[P_LIN1_B], t512, 1024, 512, 1);
  linear_kernel<<<dim3((BATCH * 256 + 255) / 256), 256, 0, stream>>>(
      t512, (const float*)d_in[P_LIN2_W], (const float*)d_in[P_LIN2_B], t256, 512, 256, 1);
  linear_kernel<<<dim3((BATCH * 40 + 255) / 256), 256, 0, stream>>>(
      t256, (const float*)d_in[P_LIN3_W], (const float*)d_in[P_LIN3_B], out, 256, 40, 0);
}
